// IE_Model_CLDR_CLNER_83743272337430
// MI455X (gfx1250) — matrix operand order repaired and checked by reading, not compiled, not run on hardware
//
#include <hip/hip_runtime.h>
#include <hip/hip_bf16.h>
#include <math.h>

typedef __attribute__((ext_vector_type(16))) _Float16 v16h;
typedef __attribute__((ext_vector_type(8)))  _Float16 v8h;
typedef __attribute__((ext_vector_type(8)))  float    v8f;
typedef __attribute__((ext_vector_type(4)))  int      v4i;

#define S_LEN 2048
#define D_DIM 1024
#define H_DIM 512
#define G_DIM 2048   /* 4*H */
#define T_TAG 5
#define KP_RC 1056   /* 1029 padded up to multiple of 32 */

#if __has_builtin(__builtin_amdgcn_global_load_async_to_lds_b128) && \
    __has_builtin(__builtin_amdgcn_s_wait_asynccnt)
#define USE_ASYNC_LDS 1
#else
#define USE_ASYNC_LDS 0
#endif

/* typed, address-space-qualified casts for the async copy builtin */
#define AS1V4(p) ((__attribute__((address_space(1))) v4i*)(p))
#define AS3V4(p) ((__attribute__((address_space(3))) v4i*)(p))

__device__ __forceinline__ float sigf(float x) { return 1.0f / (1.0f + __expf(-x)); }

/* ---------------- elementwise converts ---------------- */
__global__ void k_f32_to_f16(const float* __restrict__ src, _Float16* __restrict__ dst, long n) {
  long i = (long)blockIdx.x * blockDim.x + threadIdx.x;
  if (i < n) dst[i] = (_Float16)src[i];
}

__global__ void k_f32_to_f16_pad(const float* __restrict__ src, _Float16* __restrict__ dst,
                                 int rows, int srcK, int dstK) {
  long i = (long)blockIdx.x * blockDim.x + threadIdx.x;
  long n = (long)rows * dstK;
  if (i >= n) return;
  int r = (int)(i / dstK), c = (int)(i % dstK);
  dst[i] = (c < srcK) ? (_Float16)src[(long)r * srcK + c] : (_Float16)0.0f;
}

/* ---------------- WMMA GEMM: C[M,N] = A[M,K] * op(B) + bias[N] ----------------
 * TRANSB=1: B stored as W[N,K] row-major (computes A @ W.T)
 * TRANSB=0: B stored as B[K,N] row-major (computes A @ B)
 * Block = 4 waves = 128(M) x 64(N) C-tile; each wave owns 32(M) x 64(N):
 * two A fragments x four B fragments = 8 WMMAs per 32-wide k-step, against a
 * single set of B-fragment LDS reads (2x matrix-op density vs 16-row waves).
 * Shared 64x32 B-tile staged in LDS (double-buffered) via async global->LDS
 * copies (ASYNCcnt); the final k-step is peeled so the steady-state loop has
 * no conditionals: stage(next) -> wait(2) -> barrier -> preload A(next) ->
 * 8x WMMA(cur) -> barrier.
 * REQUIRES: M % 128 == 0, N % 64 == 0, K % 32 == 0 (true for all uses here). */
template <int TRANSB>
__global__ __launch_bounds__(128)
void k_gemm_wmma(const _Float16* __restrict__ A, const _Float16* __restrict__ B,
                 const float* __restrict__ bias, float* __restrict__ C,
                 int M, int N, int K, int lda, int ldb, int ldc) {
  __shared__ _Float16 ldsB[2][64 * 32];   /* [buf][n_local*32 + kk] */

  const int tidx  = threadIdx.x;
  const int lane  = tidx & 31;
  const int wave  = tidx >> 5;
  const int tileM = blockIdx.y * 128 + wave * 32;
  const int tileN = blockIdx.x * 64;
  const int nl    = lane & 15;          /* N (B/C/D role) / M (A role) in subtile */
  const int s     = lane >> 4;          /* K / M-half selector                     */

  /* staging roles: 128 threads cover 64 rows x 2 half-rows of 16 halves (32B) */
  const int srow  = tidx >> 1;
  const int shalf = tidx & 1;

  const _Float16* Arow0 = A + (long)(tileM + nl) * lda;
  const _Float16* Arow1 = Arow0 + (long)16 * lda;

  v8f acc[2][4] = {};

  /* ---- stage B tile for a given k-block into LDS buffer ---- */
  auto stageB = [&](int buf, int kb) {
    _Float16* dst = &ldsB[buf][srow * 32 + shalf * 16];
    if (TRANSB) {
      const _Float16* src = B + (long)(tileN + srow) * ldb + kb + shalf * 16;
#if USE_ASYNC_LDS
      __builtin_amdgcn_global_load_async_to_lds_b128(AS1V4(src),     AS3V4(dst),     0, 0);
      __builtin_amdgcn_global_load_async_to_lds_b128(AS1V4(src + 8), AS3V4(dst + 8), 0, 0);
#else
      v8h x0 = *(const v8h*)(src);
      v8h x1 = *(const v8h*)(src + 8);
      *(v8h*)(dst)     = x0;
      *(v8h*)(dst + 8) = x1;
#endif
    } else {
#pragma unroll
      for (int j = 0; j < 16; ++j)
        dst[j] = B[(long)(kb + shalf * 16 + j) * ldb + tileN + srow];
    }
  };

  auto loadA = [&](const _Float16* Arow, int kb) -> v16h {
    v8h a0 = *(const v8h*)(Arow + kb + s * 16);
    v8h a1 = *(const v8h*)(Arow + kb + s * 16 + 8);
    v16h a;
#pragma unroll
    for (int u = 0; u < 8; ++u) { a[u] = a0[u]; a[u + 8] = a1[u]; }
    return a;
  };

  /* ---- compute one 32-wide k-step from LDS buffer `buf` ---- */
  auto compute = [&](int buf, v16h aw0, v16h aw1) {
#pragma unroll
    for (int t = 0; t < 4; ++t) {
      const _Float16* bp = &ldsB[buf][(16 * t + nl) * 32 + s * 16];
      v8h b0 = *(const v8h*)(bp);
      v8h b1 = *(const v8h*)(bp + 8);
      v16h b;
#pragma unroll
      for (int u = 0; u < 8; ++u) { b[u] = b0[u]; b[u + 8] = b1[u]; }
      acc[0][t] = __builtin_amdgcn_wmma_f32_16x16x32_f16(
          false, aw0, false, b, (short)0, acc[0][t], false, false);
      acc[1][t] = __builtin_amdgcn_wmma_f32_16x16x32_f16(
          false, aw1, false, b, (short)0, acc[1][t], false, false);
    }
  };

  /* ---- pipeline prologue ---- */
  stageB(0, 0);
  v16h aw0 = loadA(Arow0, 0);
  v16h aw1 = loadA(Arow1, 0);

  int buf = 0;
  int kb  = 0;
  /* ---- steady state: unconditional stage + preload ---- */
  for (; kb + 32 < K; kb += 32) {
    stageB(buf ^ 1, kb + 32);           /* next tile in flight */
#if USE_ASYNC_LDS
    __builtin_amdgcn_s_wait_asynccnt(2);    /* only the 2 new ops may remain */
#endif
    __syncthreads();                    /* current buffer visible to all waves */
    v16h a0n = loadA(Arow0, kb + 32);   /* issued early; consumed next iter */
    v16h a1n = loadA(Arow1, kb + 32);
    compute(buf, aw0, aw1);
    aw0 = a0n; aw1 = a1n;
    __syncthreads();                    /* reads done before buffer reuse */
    buf ^= 1;
  }
  /* ---- peeled final k-step ---- */
#if USE_ASYNC_LDS
  __builtin_amdgcn_s_wait_asynccnt(0);
#endif
  __syncthreads();
  compute(buf, aw0, aw1);

  /* ---- epilogue ---- */
#pragma unroll
  for (int h = 0; h < 2; ++h) {
#pragma unroll
    for (int t = 0; t < 4; ++t) {
      const int n = tileN + 16 * t + nl;
      const float bv = bias ? bias[n] : 0.0f;
#pragma unroll
      for (int r = 0; r < 8; ++r) {
        const int m = tileM + h * 16 + r + s * 8;
        C[(long)m * ldc + n] = acc[h][t][r] + bv;
      }
    }
  }
}

/* ---------------- sequential LSTM scan (one block per direction) ----------------
 * gg = gx[t] + h @ Whh.T ; gate split [i|f|g|o]; writes h (f16) into concat buffer */
__global__ __launch_bounds__(1024)
void k_lstm_scan(const float* __restrict__ gx0, const float* __restrict__ gx1,
                 const _Float16* __restrict__ Whh0, const _Float16* __restrict__ Whh1,
                 _Float16* __restrict__ hcat16) {
  const int dir = blockIdx.x;
  const float*    gx  = dir ? gx1  : gx0;
  const _Float16* Whh = dir ? Whh1 : Whh0;
  const int tid = threadIdx.x;

  __shared__ float hS[H_DIM];
  __shared__ float cS[H_DIM];
  __shared__ float gS[G_DIM];
  if (tid < H_DIM) { hS[tid] = 0.0f; cS[tid] = 0.0f; }
  __syncthreads();

  const int j0 = tid, j1 = tid + 1024;
  const _Float16* w0 = Whh + (long)j0 * H_DIM;
  const _Float16* w1 = Whh + (long)j1 * H_DIM;

  for (int step = 0; step < S_LEN; ++step) {
    const int t = dir ? (S_LEN - 1 - step) : step;
    if (step + 1 < S_LEN) {
      const int tn = dir ? (S_LEN - 2 - step) : (step + 1);
      __builtin_prefetch((const void*)(gx + (long)tn * G_DIM + j0), 0, 0);
    }
    float a0 = gx[(long)t * G_DIM + j0];
    float a1 = gx[(long)t * G_DIM + j1];
    for (int kb = 0; kb < H_DIM; kb += 8) {
      v8h wa = *(const v8h*)(w0 + kb);
      v8h wb = *(const v8h*)(w1 + kb);
#pragma unroll
      for (int u = 0; u < 8; ++u) {
        const float hv = hS[kb + u];
        a0 += hv * (float)wa[u];
        a1 += hv * (float)wb[u];
      }
    }
    gS[j0] = a0; gS[j1] = a1;
    __syncthreads();
    if (tid < H_DIM) {
      const float ig = sigf(gS[tid]);
      const float fg = sigf(gS[tid + 512]);
      const float gg = tanhf(gS[tid + 1024]);
      const float og = sigf(gS[tid + 1536]);
      const float c  = fg * cS[tid] + ig * gg;
      const float h  = og * tanhf(c);
      cS[tid] = c; hS[tid] = h;
      hcat16[(long)t * (2 * H_DIM) + dir * H_DIM + tid] = (_Float16)h;
    }
    __syncthreads();
  }
}

/* ---------------- fc2: logits = fc1out @ fc2_w.T + b ; write to ws + d_out ------ */
__global__ void k_fc2(const float* __restrict__ fc1out, const float* __restrict__ w,
                      const float* __restrict__ b, float* __restrict__ logits,
                      float* __restrict__ out) {
  int i = blockIdx.x * blockDim.x + threadIdx.x;
  if (i >= S_LEN * T_TAG) return;
  const int srow = i / T_TAG, t = i % T_TAG;
  float acc = b[t];
  const float* x  = fc1out + (long)srow * 64;
  const float* wr = w + t * 64;
#pragma unroll 8
  for (int k = 0; k < 64; ++k) acc += x[k] * wr[k];
  logits[i] = acc;
  out[i]    = acc;   /* ner_model_output (S,1,T) is flat-identical to logits */
}

/* ---------------- Viterbi (T=5, trivial; single thread) ---------------- */
__global__ void k_viterbi(const float* __restrict__ logits, const float* __restrict__ start,
                          const float* __restrict__ trans, const float* __restrict__ endv,
                          int* __restrict__ bptr, int* __restrict__ tags) {
  if (threadIdx.x != 0 || blockIdx.x != 0) return;
  float score[T_TAG];
  for (int j = 0; j < T_TAG; ++j) score[j] = start[j] + logits[j];
  for (int t = 1; t < S_LEN; ++t) {
    float ns[T_TAG];
    for (int j = 0; j < T_TAG; ++j) {
      float best = score[0] + trans[0 * T_TAG + j]; int bi = 0;
      for (int i = 1; i < T_TAG; ++i) {
        const float v = score[i] + trans[i * T_TAG + j];
        if (v > best) { best = v; bi = i; }
      }
      ns[j] = best + logits[t * T_TAG + j];
      bptr[(t - 1) * T_TAG + j] = bi;
    }
    for (int j = 0; j < T_TAG; ++j) score[j] = ns[j];
  }
  int last = 0; float bv = score[0] + endv[0];
  for (int j = 1; j < T_TAG; ++j) { const float v = score[j] + endv[j]; if (v > bv) { bv = v; last = j; } }
  tags[S_LEN - 1] = last;
  int cur = last;
  for (int t = S_LEN - 2; t >= 0; --t) { cur = bptr[t * T_TAG + cur]; tags[t] = cur; }
}

/* ---------------- rc_in = [one_hot(tags) | embRE | zero-pad] as f16 ------------- */
__global__ void k_build_rcin(const float* __restrict__ embRE, const int* __restrict__ tags,
                             _Float16* __restrict__ rcin) {
  long i = (long)blockIdx.x * blockDim.x + threadIdx.x;
  if (i >= (long)S_LEN * KP_RC) return;
  const int r = (int)(i / KP_RC), c = (int)(i % KP_RC);
  float v;
  if (c < T_TAG)                 v = (tags[r] == c) ? 1.0f : 0.0f;
  else if (c < T_TAG + D_DIM)    v = embRE[(long)r * D_DIM + (c - T_TAG)];
  else                           v = 0.0f;
  rcin[i] = (_Float16)v;
}

/* ---------------- row gather (f16) ---------------- */
__global__ void k_gather_rows(const _Float16* __restrict__ src, const int* __restrict__ idx,
                              _Float16* __restrict__ dst, int rows, int cols) {
  long i = (long)blockIdx.x * blockDim.x + threadIdx.x;
  if (i >= (long)rows * cols) return;
  const int r = (int)(i / cols), c = (int)(i % cols);
  dst[i] = src[(long)idx[r] * cols + c];
}

/* ---------------- biaffine linear terms ---------------- */
__global__ void k_lin(const float* __restrict__ head, const float* __restrict__ tail,
                      const float* __restrict__ lw, const float* __restrict__ lb,
                      float* __restrict__ lin, float* __restrict__ lint) {
  const int i = blockIdx.x * blockDim.x + threadIdx.x;
  if (i < 128) {
    float a = lb[0];
    const float* x = head + (long)i * D_DIM;
    for (int k = 0; k < D_DIM; ++k) a += x[k] * lw[k];
    lin[i] = a;
  } else if (i < 256) {
    const int p = i - 128;
    float a = 0.0f;
    const float* x = tail + (long)p * D_DIM;
    for (int k = 0; k < D_DIM; ++k) a += x[k] * lw[D_DIM + k];
    lint[p] = a;
  }
}

__global__ void k_probs(const float* __restrict__ bil, const float* __restrict__ lin,
                        const float* __restrict__ lint, float* __restrict__ out) {
  const int i = blockIdx.x * blockDim.x + threadIdx.x;
  if (i >= 128 * 128) return;
  const int p = i >> 7, q = i & 127;
  out[i] = sigf(bil[i] + lin[p] + lint[q]);
}

/* ================================ host side ================================ */
static inline void launch_gemm(int transB, const _Float16* A, const _Float16* B,
                               const float* bias, float* C, int M, int N, int K,
                               int lda, int ldb, int ldc, hipStream_t s) {
  dim3 grid(N / 64, M / 128), block(128);
  if (transB) k_gemm_wmma<1><<<grid, block, 0, s>>>(A, B, bias, C, M, N, K, lda, ldb, ldc);
  else        k_gemm_wmma<0><<<grid, block, 0, s>>>(A, B, bias, C, M, N, K, lda, ldb, ldc);
}

static inline void launch_cvt(const float* src, _Float16* dst, long n, hipStream_t s) {
  k_f32_to_f16<<<dim3((unsigned)((n + 255) / 256)), dim3(256), 0, s>>>(src, dst, n);
}

extern "C" void kernel_launch(void* const* d_in, const int* in_sizes, int n_in,
                              void* d_out, int out_size, void* d_ws, size_t ws_size,
                              hipStream_t stream) {
  (void)in_sizes; (void)n_in; (void)out_size; (void)ws_size;

  const float* embNER   = (const float*)d_in[0];
  const float* embRE    = (const float*)d_in[1];
  const float* nerWih0  = (const float*)d_in[2];
  const float* nerWhh0  = (const float*)d_in[3];
  const float* ner_b0   = (const float*)d_in[4];
  const float* nerWih1  = (const float*)d_in[5];
  const float* nerWhh1  = (const float*)d_in[6];
  const float* ner_b1   = (const float*)d_in[7];
  const float* fc1_w    = (const float*)d_in[8];
  const float* fc1_b    = (const float*)d_in[9];
  const float* fc2_w    = (const float*)d_in[10];
  const float* fc2_b    = (const float*)d_in[11];
  const float* crfStart = (const float*)d_in[12];
  const float* crfTrans = (const float*)d_in[13];
  const float* crfEnd   = (const float*)d_in[14];
  const float* rcWih0   = (const float*)d_in[15];
  const float* rcWhh0   = (const float*)d_in[16];
  const float* rc_b0    = (const float*)d_in[17];
  const float* rcWih1   = (const float*)d_in[18];
  const float* rcWhh1   = (const float*)d_in[19];
  const float* rc_b1    = (const float*)d_in[20];
  const float* head_w   = (const float*)d_in[21];
  const float* tail_w   = (const float*)d_in[22];
  const float* biaf_u   = (const float*)d_in[23];
  const float* biaf_lw  = (const float*)d_in[24];
  const float* biaf_lb  = (const float*)d_in[25];
  const int*   drugIdx  = (const int*)d_in[26];
  const int*   effIdx   = (const int*)d_in[27];
  float* out = (float*)d_out;

  /* ---- workspace layout (deterministic bump allocator, 256B aligned) ---- */
  size_t off = 0;
  auto take = [&](size_t bytes) -> void* {
    void* p = (char*)d_ws + off;
    off = (off + bytes + 255) & ~(size_t)255;
    return p;
  };
  _Float16* nerWih0_16 = (_Float16*)take(2ull * G_DIM * D_DIM * 2);
  _Float16* nerWhh0_16 = (_Float16*)take(2ull * G_DIM * H_DIM * 2);
  _Float16* nerWih1_16 = (_Float16*)take(2ull * G_DIM * D_DIM * 2);
  _Float16* nerWhh1_16 = (_Float16*)take(2ull * G_DIM * H_DIM * 2);
  _Float16* rcWih0_16  = (_Float16*)take(2ull * G_DIM * KP_RC * 2);
  _Float16* rcWhh0_16  = (_Float16*)take(2ull * G_DIM * H_DIM * 2);
  _Float16* rcWih1_16  = (_Float16*)take(2ull * G_DIM * D_DIM * 2);
  _Float16* rcWhh1_16  = (_Float16*)take(2ull * G_DIM * H_DIM * 2);
  _Float16* fc1w16     = (_Float16*)take(64ull * D_DIM * 2);
  _Float16* headw16    = (_Float16*)take((size_t)D_DIM * D_DIM * 2);
  _Float16* tailw16    = (_Float16*)take((size_t)D_DIM * D_DIM * 2);
  _Float16* biafu16    = (_Float16*)take((size_t)D_DIM * D_DIM * 2);
  _Float16* xner16     = (_Float16*)take((size_t)S_LEN * D_DIM * 2);
  float*    gx0        = (float*)take((size_t)S_LEN * G_DIM * 4);
  float*    gx1        = (float*)take((size_t)S_LEN * G_DIM * 4);
  _Float16* hcat16     = (_Float16*)take((size_t)S_LEN * D_DIM * 2);
  float*    fc1out     = (float*)take((size_t)S_LEN * 64 * 4);
  float*    logits     = (float*)take((size_t)S_LEN * T_TAG * 4);
  int*      bptr       = (int*)take((size_t)(S_LEN - 1) * T_TAG * 4);
  int*      tags       = (int*)take((size_t)S_LEN * 4);
  _Float16* rcin16     = (_Float16*)take((size_t)S_LEN * KP_RC * 2);
  _Float16* headA16    = (_Float16*)take(128ull * D_DIM * 2);
  _Float16* tailA16    = (_Float16*)take(128ull * D_DIM * 2);
  float*    head32     = (float*)take(128ull * D_DIM * 4);
  float*    tail32     = (float*)take(128ull * D_DIM * 4);
  _Float16* head16     = (_Float16*)take(128ull * D_DIM * 2);
  _Float16* tail16     = (_Float16*)take(128ull * D_DIM * 2);
  float*    tmp32      = (float*)take(128ull * D_DIM * 4);
  _Float16* tmp16      = (_Float16*)take(128ull * D_DIM * 2);
  float*    bil32      = (float*)take(128ull * 128 * 4);
  float*    linv       = (float*)take(128ull * 4);
  float*    lintv      = (float*)take(128ull * 4);

  /* ---- converts (re-run every call; deterministic) ---- */
  launch_cvt(embNER,  xner16,     (long)S_LEN * D_DIM, stream);
  launch_cvt(nerWih0, nerWih0_16, 2L * G_DIM * D_DIM,  stream);
  launch_cvt(nerWhh0, nerWhh0_16, 2L * G_DIM * H_DIM,  stream);
  launch_cvt(nerWih1, nerWih1_16, 2L * G_DIM * D_DIM,  stream);
  launch_cvt(nerWhh1, nerWhh1_16, 2L * G_DIM * H_DIM,  stream);
  launch_cvt(rcWhh0,  rcWhh0_16,  2L * G_DIM * H_DIM,  stream);
  launch_cvt(rcWih1,  rcWih1_16,  2L * G_DIM * D_DIM,  stream);
  launch_cvt(rcWhh1,  rcWhh1_16,  2L * G_DIM * H_DIM,  stream);
  launch_cvt(fc1_w,   fc1w16,     64L * D_DIM,         stream);
  launch_cvt(head_w,  headw16,    (long)D_DIM * D_DIM, stream);
  launch_cvt(tail_w,  tailw16,    (long)D_DIM * D_DIM, stream);
  launch_cvt(biaf_u,  biafu16,    (long)D_DIM * D_DIM, stream);
  { long n = 2L * G_DIM * KP_RC;
    k_f32_to_f16_pad<<<dim3((unsigned)((n + 255) / 256)), dim3(256), 0, stream>>>(
        rcWih0, rcWih0_16, 2 * G_DIM, T_TAG + D_DIM, KP_RC); }

  /* ---- NER bilstm, layer 0 ---- */
  launch_gemm(1, xner16, nerWih0_16,                     ner_b0,          gx0, S_LEN, G_DIM, D_DIM, D_DIM, D_DIM, G_DIM, stream);
  launch_gemm(1, xner16, nerWih0_16 + (long)G_DIM*D_DIM, ner_b0 + G_DIM,  gx1, S_LEN, G_DIM, D_DIM, D_DIM, D_DIM, G_DIM, stream);
  k_lstm_scan<<<dim3(2), dim3(1024), 0, stream>>>(gx0, gx1, nerWhh0_16, nerWhh0_16 + (long)G_DIM*H_DIM, hcat16);
  /* ---- NER bilstm, layer 1 ---- */
  launch_gemm(1, hcat16, nerWih1_16,                     ner_b1,          gx0, S_LEN, G_DIM, D_DIM, D_DIM, D_DIM, G_DIM, stream);
  launch_gemm(1, hcat16, nerWih1_16 + (long)G_DIM*D_DIM, ner_b1 + G_DIM,  gx1, S_LEN, G_DIM, D_DIM, D_DIM, D_DIM, G_DIM, stream);
  k_lstm_scan<<<dim3(2), dim3(1024), 0, stream>>>(gx0, gx1, nerWhh1_16, nerWhh1_16 + (long)G_DIM*H_DIM, hcat16);

  /* ---- FC head + CRF ---- */
  launch_gemm(1, hcat16, fc1w16, fc1_b, fc1out, S_LEN, 64, D_DIM, D_DIM, D_DIM, 64, stream);
  k_fc2<<<dim3((S_LEN * T_TAG + 255) / 256), dim3(256), 0, stream>>>(fc1out, fc2_w, fc2_b, logits, out);
  k_viterbi<<<dim3(1), dim3(32), 0, stream>>>(logits, crfStart, crfTrans, crfEnd, bptr, tags);

  /* ---- RE input + bilstm ---- */
  { long n = (long)S_LEN * KP_RC;
    k_build_rcin<<<dim3((unsigned)((n + 255) / 256)), dim3(256), 0, stream>>>(embRE, tags, rcin16); }
  launch_gemm(1, rcin16, rcWih0_16,                     rc_b0,          gx0, S_LEN, G_DIM, KP_RC, KP_RC, KP_RC, G_DIM, stream);
  launch_gemm(1, rcin16, rcWih0_16 + (long)G_DIM*KP_RC, rc_b0 + G_DIM,  gx1, S_LEN, G_DIM, KP_RC, KP_RC, KP_RC, G_DIM, stream);
  k_lstm_scan<<<dim3(2), dim3(1024), 0, stream>>>(gx0, gx1, rcWhh0_16, rcWhh0_16 + (long)G_DIM*H_DIM, hcat16);
  launch_gemm(1, hcat16, rcWih1_16,                     rc_b1,          gx0, S_LEN, G_DIM, D_DIM, D_DIM, D_DIM, G_DIM, stream);
  launch_gemm(1, hcat16, rcWih1_16 + (long)G_DIM*D_DIM, rc_b1 + G_DIM,  gx1, S_LEN, G_DIM, D_DIM, D_DIM, D_DIM, G_DIM, stream);
  k_lstm_scan<<<dim3(2), dim3(1024), 0, stream>>>(gx0, gx1, rcWhh1_16, rcWhh1_16 + (long)G_DIM*H_DIM, hcat16);

  /* ---- biaffine scoring ---- */
  { long n = 128L * D_DIM;
    k_gather_rows<<<dim3((unsigned)((n + 255) / 256)), dim3(256), 0, stream>>>(hcat16, drugIdx, headA16, 128, D_DIM);
    k_gather_rows<<<dim3((unsigned)((n + 255) / 256)), dim3(256), 0, stream>>>(hcat16, effIdx,  tailA16, 128, D_DIM); }
  launch_gemm(1, headA16, headw16, nullptr, head32, 128, D_DIM, D_DIM, D_DIM, D_DIM, D_DIM, stream);
  launch_gemm(1, tailA16, tailw16, nullptr, tail32, 128, D_DIM, D_DIM, D_DIM, D_DIM, D_DIM, stream);
  launch_cvt(head32, head16, 128L * D_DIM, stream);
  launch_cvt(tail32, tail16, 128L * D_DIM, stream);
  launch_gemm(0, head16, biafu16, nullptr, tmp32, 128, D_DIM, D_DIM, D_DIM, D_DIM, D_DIM, stream);  /* head @ U */
  launch_cvt(tmp32, tmp16, 128L * D_DIM, stream);
  launch_gemm(1, tmp16, tail16, nullptr, bil32, 128, 128, D_DIM, D_DIM, D_DIM, 128, stream);        /* @ tail.T */
  k_lin<<<dim3(1), dim3(256), 0, stream>>>(head32, tail32, biaf_lw, biaf_lb, linv, lintv);
  k_probs<<<dim3((128 * 128 + 255) / 256), dim3(256), 0, stream>>>(bil32, linv, lintv, out + (long)S_LEN * T_TAG);
}
